// RotateLinearAWQMarlinInt4_601295421926
// MI455X (gfx1250) — matrix operand order repaired and checked by reading, not compiled, not run on hardware
//
#include <hip/hip_runtime.h>

// Problem constants (fixed by the reference setup)
#define M_DIM   32
#define K_DIM   8192
#define N_DIM   28672
#define NB      (N_DIM / 8)     // 3584 dwords per qweight row
#define QGS_C   128             // quant group size along K
#define KROT_C  8
#define GROUP_C 128

// GEMM tiling
#define SPLITK   4
#define KCHUNK   (K_DIM / SPLITK)   // 2048
#define WAVES    4
#define THREADS  (WAVES * 32)       // 128, wave32
#define NTILE_WG 128                // 4 waves x 32 columns
#define NDW      (NTILE_WG / 8)     // 16 dwords of qweight per k-row
#define KSTAGE   128                // one quant group per LDS stage
#define QBUF_DW  (KSTAGE * NDW)     // 2048 dwords = 8 KB
#define ABUF_H   (M_DIM * KSTAGE)   // 4096 halves = 8 KB

typedef __attribute__((ext_vector_type(16))) _Float16 v16h;
typedef __attribute__((ext_vector_type(8)))  _Float16 v8h;
typedef __attribute__((ext_vector_type(2)))  _Float16 v2h;
typedef __attribute__((ext_vector_type(8)))  float    v8f;

union BFrag { v16h h; unsigned u[8]; };
union H2U   { unsigned u; v2h h; };

__device__ __forceinline__ v16h cat8(v8h lo, v8h hi) {
  return __builtin_shufflevector(lo, hi, 0,1,2,3,4,5,6,7,8,9,10,11,12,13,14,15);
}

// CDNA5 async copy: 16B global -> LDS per lane, tracked by ASYNCcnt.
// GVS addressing: 64-bit SGPR base + 32-bit per-lane byte offset.
__device__ __forceinline__ void async_b128(unsigned lds_off, unsigned gl_off,
                                           const void* base) {
  asm volatile("global_load_async_to_lds_b128 %0, %1, %2"
               :: "v"(lds_off), "v"(gl_off), "s"(base)
               : "memory");
}
__device__ __forceinline__ void wait_async0() {
  asm volatile("s_wait_asynccnt 0" ::: "memory");
}

// ---------------------------------------------------------------------------
// Kernel 1: Givens rotations + channel scaling.  One block per K-group.
// ---------------------------------------------------------------------------
__global__ __launch_bounds__(128) void rotate_kernel(
    const _Float16* __restrict__ x,
    const _Float16* __restrict__ theta,
    const int*      __restrict__ pairs,
    const _Float16* __restrict__ cscale,
    _Float16*       __restrict__ xr)
{
  __shared__ float xs[M_DIM][GROUP_C];
  const int g = blockIdx.x;        // K-group index, 0..63
  const int t = threadIdx.x;       // 0..127

  #pragma unroll
  for (int m = 0; m < M_DIM; ++m)
    xs[m][t] = (float)x[(size_t)m * K_DIM + (size_t)g * GROUP_C + t];
  __syncthreads();

  for (int k = 0; k < KROT_C; ++k) {
    if (t < GROUP_C / 2) {
      const int i = pairs[k * GROUP_C + 2 * t];
      const int j = pairs[k * GROUP_C + 2 * t + 1];
      const float th = (float)theta[(size_t)k * (K_DIM / 2) + g * (GROUP_C / 2) + t];
      const float c = cosf(th);
      const float s = sinf(th);
      #pragma unroll
      for (int m = 0; m < M_DIM; ++m) {
        const float xi = xs[m][i];
        const float xj = xs[m][j];
        xs[m][i] = c * xi - s * xj;
        xs[m][j] = s * xi + c * xj;
      }
    }
    __syncthreads();
  }

  const float cs = (float)cscale[g * GROUP_C + t];
  #pragma unroll
  for (int m = 0; m < M_DIM; ++m)
    xr[(size_t)m * K_DIM + (size_t)g * GROUP_C + t] = (_Float16)(xs[m][t] * cs);
}

// ---------------------------------------------------------------------------
// Kernel 2: initialize output with broadcast bias (float accumulation target).
// ---------------------------------------------------------------------------
__global__ __launch_bounds__(256) void init_out_kernel(
    const _Float16* __restrict__ bias, float* __restrict__ out)
{
  const size_t idx = (size_t)blockIdx.x * 256 + threadIdx.x;   // < M*N exactly
  out[idx] = (float)bias[idx % N_DIM];
}

// ---------------------------------------------------------------------------
// Kernel 3: int4 dequant + WMMA GEMM, double-buffered async-to-LDS staging,
// split-K with atomic accumulation.
// ---------------------------------------------------------------------------
__global__ __launch_bounds__(THREADS) void gemm_i4_kernel(
    const _Float16* __restrict__ xr,
    const int*      __restrict__ qweight,
    const int*      __restrict__ qzeros,
    const _Float16* __restrict__ scales,
    float*          __restrict__ out)
{
  __shared__ unsigned lds_q[2 * QBUF_DW];     // 2 x 8 KB: int4 weight tiles
  __shared__ _Float16 lds_a[2 * ABUF_H];      // 2 x 8 KB: activation tiles

  const int t     = threadIdx.x;
  const int wave  = t >> 5;
  const int lane  = t & 31;
  const int lhalf = lane >> 4;                // 0 or 1 (WMMA lane half)
  const int l16   = lane & 15;

  const int n0   = blockIdx.x * NTILE_WG;
  const int kbeg = blockIdx.y * KCHUNK;

  const int nloc0 = wave * 32 + l16;          // local column of n-tile 0
  const int nloc1 = nloc0 + 16;               // n-tile 1
  const int ncol0 = n0 + nloc0;
  const int ncol1 = n0 + nloc1;
  const int ndw0  = nloc0 >> 3;
  const int ndw1  = nloc1 >> 3;
  const int shn   = (l16 & 7) * 4;            // nibble shift for this column

  // LDS byte offsets (low 32 bits of the shared-aperture flat address are the
  // workgroup-relative LDS offset, which is what async-to-LDS VDST expects).
  const unsigned q_lds0 = (unsigned)(size_t)(&lds_q[0]);
  const unsigned a_lds0 = (unsigned)(size_t)(&lds_a[0]);

  v8f acc00 = {}, acc01 = {}, acc10 = {}, acc11 = {};

  // Stage one 128-k slice (one quant group) into buffer `buf`:
  //   qweight: KSTAGE x NDW dwords as 512 16B chunks (4 per thread)
  //   activations: M_DIM x KSTAGE halves as 512 16B chunks (4 per thread)
  auto stage = [&](int buf, int kg) {
    const unsigned qb = q_lds0 + (unsigned)buf * (QBUF_DW * 4u);
    const unsigned ab = a_lds0 + (unsigned)buf * (ABUF_H * 2u);
    #pragma unroll
    for (int r = 0; r < 4; ++r) {
      const int idx = r * THREADS + t;
      // qweight chunk: row kk, dwords cd*4..cd*4+3 (64B-contiguous rows)
      const int kk = idx >> 2, cd = idx & 3;
      const unsigned goq =
          (unsigned)(((size_t)(kg + kk) * NB + (size_t)(n0 >> 3) + cd * 4) * 4);
      async_b128(qb + (unsigned)idx * 16u, goq, qweight);
      // activation chunk: row m, halves c*8..c*8+7
      const int m = idx >> 4, c = idx & 15;
      const unsigned goa = (unsigned)(((size_t)m * K_DIM + kg + c * 8) * 2);
      async_b128(ab + (unsigned)idx * 16u, goa, xr);
    }
  };

  stage(0, kbeg);
  wait_async0();
  __syncthreads();

  int buf = 0;
  for (int ks = 0; ks < KCHUNK; ks += KSTAGE) {
    const int kg = kbeg + ks;

    // Prefetch next slice into the other buffer while computing this one.
    if (ks + KSTAGE < KCHUNK) stage(buf ^ 1, kg + KSTAGE);

    const unsigned*  lq = &lds_q[(size_t)buf * QBUF_DW];
    const _Float16*  ax = &lds_a[(size_t)buf * ABUF_H];

    // Per-group dequant constants (KSTAGE == QGS): w = s*hmagic - s*(1024+z)
    const int g = kg / QGS_C;
    const unsigned zw0 = (unsigned)qzeros[(size_t)g * NB + (ncol0 >> 3)];
    const unsigned zw1 = (unsigned)qzeros[(size_t)g * NB + (ncol1 >> 3)];
    const float z0 = (float)((zw0 >> shn) & 15u);
    const float z1 = (float)((zw1 >> shn) & 15u);
    const _Float16 s0 = scales[(size_t)g * N_DIM + ncol0];
    const _Float16 s1 = scales[(size_t)g * N_DIM + ncol1];
    const _Float16 c0 = (_Float16)(-(float)s0 * (1024.0f + z0));
    const _Float16 c1 = (_Float16)(-(float)s1 * (1024.0f + z1));
    const v2h sp0 = { s0, s0 }, cp0 = { c0, c0 };
    const v2h sp1 = { s1, s1 }, cp1 = { c1, c1 };

    #pragma unroll
    for (int kt = 0; kt < KSTAGE; kt += 32) {
      // A fragments (two M tiles) from LDS, matching the B k order:
      // lanes 0-15: K 0..15; lanes 16-31: K 16..31.
      const _Float16* ap0 = &ax[(size_t)l16 * KSTAGE + kt + lhalf * 16];
      const _Float16* ap1 = &ax[(size_t)(16 + l16) * KSTAGE + kt + lhalf * 16];
      const v16h a0 = cat8(*(const v8h*)ap0, *(const v8h*)(ap0 + 8));
      const v16h a1 = cat8(*(const v8h*)ap1, *(const v8h*)(ap1 + 8));

      // B fragments: lanes 0-15 hold K 0..15, lanes 16-31 hold K 16..31;
      // half-pair v holds rows (2v, 2v+1). Magic 0x6400 => f16 value 1024+nib.
      BFrag b0, b1;
      #pragma unroll
      for (int v = 0; v < 8; ++v) {
        const int krow = kt + lhalf * 16 + 2 * v;
        const unsigned lo0 = lq[krow * NDW + ndw0];
        const unsigned hi0 = lq[(krow + 1) * NDW + ndw0];
        H2U r0; r0.u = 0x64006400u | ((lo0 >> shn) & 15u) | (((hi0 >> shn) & 15u) << 16);
        H2U d0; d0.h = r0.h * sp0 + cp0;      // v_pk ops
        b0.u[v] = d0.u;
        const unsigned lo1 = lq[krow * NDW + ndw1];
        const unsigned hi1 = lq[(krow + 1) * NDW + ndw1];
        H2U r1; r1.u = 0x64006400u | ((lo1 >> shn) & 15u) | (((hi1 >> shn) & 15u) << 16);
        H2U d1; d1.h = r1.h * sp1 + cp1;
        b1.u[v] = d1.u;
      }

      acc00 = __builtin_amdgcn_wmma_f32_16x16x32_f16(false, a0, false, b0.h, (short)0, acc00, false, false);
      acc01 = __builtin_amdgcn_wmma_f32_16x16x32_f16(false, a1, false, b0.h, (short)0, acc01, false, false);
      acc10 = __builtin_amdgcn_wmma_f32_16x16x32_f16(false, a0, false, b1.h, (short)0, acc10, false, false);
      acc11 = __builtin_amdgcn_wmma_f32_16x16x32_f16(false, a1, false, b1.h, (short)0, acc11, false, false);
    }

    // Next buffer fully landed, and all waves done reading this one.
    wait_async0();
    __syncthreads();
    buf ^= 1;
  }

  // C layout: VGPR r => M = r (lanes 0-15) / 8+r (lanes 16-31), N = lane%16.
  #pragma unroll
  for (int r = 0; r < 8; ++r) {
    const int mA = lhalf * 8 + r;       // M tile 0 rows 0..15
    const int mB = 16 + mA;             // M tile 1 rows 16..31
    atomicAdd(&out[(size_t)mA * N_DIM + ncol0], acc00[r]);
    atomicAdd(&out[(size_t)mB * N_DIM + ncol0], acc01[r]);
    atomicAdd(&out[(size_t)mA * N_DIM + ncol1], acc10[r]);
    atomicAdd(&out[(size_t)mB * N_DIM + ncol1], acc11[r]);
  }
}

// ---------------------------------------------------------------------------
extern "C" void kernel_launch(void* const* d_in, const int* in_sizes, int n_in,
                              void* d_out, int out_size, void* d_ws, size_t ws_size,
                              hipStream_t stream) {
  (void)in_sizes; (void)n_in; (void)out_size; (void)ws_size;
  const _Float16* x       = (const _Float16*)d_in[0];
  const int*      qweight = (const int*)d_in[1];
  const int*      qzeros  = (const int*)d_in[2];
  const _Float16* scales  = (const _Float16*)d_in[3];
  const _Float16* bias    = (const _Float16*)d_in[4];
  const _Float16* theta   = (const _Float16*)d_in[5];
  const int*      pairs   = (const int*)d_in[6];
  const _Float16* cscale  = (const _Float16*)d_in[7];

  float*    out = (float*)d_out;
  _Float16* xr  = (_Float16*)d_ws;   // 32*8192*2 = 512 KB scratch

  rotate_kernel<<<K_DIM / GROUP_C, 128, 0, stream>>>(x, theta, pairs, cscale, xr);
  init_out_kernel<<<(M_DIM * N_DIM) / 256, 256, 0, stream>>>(bias, out);
  gemm_i4_kernel<<<dim3(N_DIM / NTILE_WG, SPLITK), THREADS, 0, stream>>>(
      xr, qweight, qzeros, scales, out);
}
